// SpatialTemporalModel_13563506721370
// MI455X (gfx1250) — matrix operand order repaired and checked by reading, not compiled, not run on hardware
//
#include <hip/hip_runtime.h>
#include <math.h>

typedef _Float16 half_t;
typedef __attribute__((ext_vector_type(16))) _Float16 v16h;
typedef __attribute__((ext_vector_type(8)))  float    v8f;

// ---------------------------------------------------------------------------
// Gated temporal conv:  out = relu((X*W1+b1) + sigmoid(X*W2+b2) + (X*W3+b3))
// X layout (bn, t, c) with c fastest -> K=3 window is a contiguous 3*CIN slice.
// One wave32 per 16-row tile; weights (f16, transposed) staged in LDS.
// ---------------------------------------------------------------------------
template<int CIN, int COUT, int COLS, int TIN, int TOUT, bool IN_F32>
__global__ __launch_bounds__(256)
void tconv_gate_kernel(const void* __restrict__ inv,
                       const float* __restrict__ w1, const float* __restrict__ b1,
                       const float* __restrict__ w2, const float* __restrict__ b2,
                       const float* __restrict__ w3, const float* __restrict__ b3,
                       half_t* __restrict__ outp, int rowTiles, int colOff)
{
    constexpr int L    = 3 * CIN;
    constexpr int LPAD = (L + 31) & ~31;
    constexpr int NTL  = COLS / 16;
    constexpr int NCH  = LPAD / 32;

    __shared__ half_t WT[3][COLS][LPAD];   // WT[w][o][k] = w[k*COUT + colOff+o]

    const int tid = threadIdx.x;
    for (int idx = tid; idx < 3 * COLS * LPAD; idx += 256) {
        int wsel = idx / (COLS * LPAD);
        int rem  = idx - wsel * (COLS * LPAD);
        int o    = rem / LPAD;
        int l    = rem - o * LPAD;
        const float* wsrc = (wsel == 0) ? w1 : ((wsel == 1) ? w2 : w3);
        WT[wsel][o][l] = (l < L) ? (half_t)wsrc[l * COUT + colOff + o] : (half_t)0.f;
    }
    __syncthreads();

    const int wave = tid >> 5;
    const int lane = tid & 31;
    const int m    = lane & 15;
    const int hi   = lane >> 4;

    const int rt = blockIdx.x * 8 + wave;
    if (rt >= rowTiles) return;

    // this lane supplies A-matrix row m of the tile
    const int  row   = rt * 16 + m;
    const int  bn    = row / TOUT;
    const int  t     = row - bn * TOUT;
    const long abase = ((long)bn * TIN + t) * CIN;

    const float*  inF = (const float*)inv;
    const half_t* inH = (const half_t*)inv;

    const v8f zero = {0.f, 0.f, 0.f, 0.f, 0.f, 0.f, 0.f, 0.f};
    v8f acc[3][NTL];
    #pragma unroll
    for (int w = 0; w < 3; ++w)
        #pragma unroll
        for (int nt = 0; nt < NTL; ++nt)
            acc[w][nt] = zero;

    #pragma unroll
    for (int kc = 0; kc < NCH; ++kc) {
        const int kbase = kc * 32;
        // A fragment (16x32 f16): element i <-> K = (i&7) + 8*hi + 16*(i>>3)
        v16h a;
        #pragma unroll
        for (int i = 0; i < 16; ++i) {
            const int k = kbase + (i & 7) + 8 * hi + ((i >> 3) << 4);
            float v = 0.f;
            if (k < L) v = IN_F32 ? inF[abase + k] : (float)inH[abase + k];
            a[i] = (half_t)v;
        }
        #pragma unroll
        for (int nt = 0; nt < NTL; ++nt) {
            #pragma unroll
            for (int w = 0; w < 3; ++w) {
                // B fragment (32x16 f16): element i <-> K = kbase + (i&7) + 8*hi + 16*(i>>3), col = m
                v16h bf;
                #pragma unroll
                for (int i = 0; i < 16; ++i)
                    bf[i] = WT[w][nt * 16 + m][kbase + (i & 7) + 8 * hi + ((i >> 3) << 4)];
                acc[w][nt] = __builtin_amdgcn_wmma_f32_16x16x32_f16(
                    false, a, false, bf, (short)0, acc[w][nt], false, false);
            }
        }
    }

    // epilogue: C/D VGPR j <-> row j + 8*hi, col = lane&15
    #pragma unroll
    for (int nt = 0; nt < NTL; ++nt) {
        const int   col = colOff + nt * 16 + m;
        const float bb1 = b1[col], bb2 = b2[col], bb3 = b3[col];
        #pragma unroll
        for (int j = 0; j < 8; ++j) {
            const long  orow = (long)rt * 16 + j + 8 * hi;
            const float c1 = acc[0][nt][j] + bb1;
            const float c2 = acc[1][nt][j] + bb2;
            const float c3 = acc[2][nt][j] + bb3;
            const float sg = 1.f / (1.f + expf(-c2));
            float r = c1 + sg + c3;
            r = r > 0.f ? r : 0.f;
            outp[orow * COUT + col] = (half_t)r;
        }
    }
}

// ---------------------------------------------------------------------------
// ChebConv K=1: per-row 16->16 linear + relu  (single 16x16x32 WMMA, K padded)
// ---------------------------------------------------------------------------
__global__ __launch_bounds__(256)
void cheb_relu_kernel(const half_t* __restrict__ in, const float* __restrict__ w,
                      const float* __restrict__ b, half_t* __restrict__ outp,
                      int rowTiles)
{
    __shared__ half_t WT[16][32];
    const int tid = threadIdx.x;
    for (int idx = tid; idx < 16 * 32; idx += 256) {
        int o = idx >> 5, l = idx & 31;
        WT[o][l] = (l < 16) ? (half_t)w[l * 16 + o] : (half_t)0.f;
    }
    __syncthreads();

    const int wave = tid >> 5, lane = tid & 31;
    const int m = lane & 15, hi = lane >> 4;
    const int rt = blockIdx.x * 8 + wave;
    if (rt >= rowTiles) return;

    const long abase = ((long)rt * 16 + m) * 16;
    v16h a;
    #pragma unroll
    for (int i = 0; i < 16; ++i) {
        const int k = (i & 7) + 8 * hi + ((i >> 3) << 4);
        a[i] = (k < 16) ? in[abase + k] : (half_t)0.f;
    }
    v16h bf;
    #pragma unroll
    for (int i = 0; i < 16; ++i)
        bf[i] = WT[m][(i & 7) + 8 * hi + ((i >> 3) << 4)];

    v8f c = {0.f, 0.f, 0.f, 0.f, 0.f, 0.f, 0.f, 0.f};
    c = __builtin_amdgcn_wmma_f32_16x16x32_f16(false, a, false, bf, (short)0, c,
                                               false, false);
    const float bias = b[m];
    #pragma unroll
    for (int j = 0; j < 8; ++j) {
        const long orow = (long)rt * 16 + j + 8 * hi;
        float v = c[j] + bias;
        outp[orow * 16 + m] = (half_t)(v > 0.f ? v : 0.f);
    }
}

// ---------------------------------------------------------------------------
// BatchNorm over node channel: stats per n over (B=8, T, 64), then apply.
// ---------------------------------------------------------------------------
__global__ __launch_bounds__(256)
void bn_stats_kernel(const half_t* __restrict__ x, float* __restrict__ meanOut,
                     float* __restrict__ rstdOut, int TT)
{
    const int n = blockIdx.x;           // 0..1999
    const int tid = threadIdx.x;
    const int per_b = TT * 64;
    const int cnt = 8 * per_b;
    float s = 0.f, ss = 0.f;
    for (int idx = tid; idx < cnt; idx += 256) {
        const int b = idx / per_b;
        const int rem = idx - b * per_b;
        const float v = (float)x[((long)(b * 2000 + n)) * per_b + rem];
        s += v; ss += v * v;
    }
    __shared__ float shs[256], shq[256];
    shs[tid] = s; shq[tid] = ss;
    __syncthreads();
    for (int off = 128; off > 0; off >>= 1) {
        if (tid < off) { shs[tid] += shs[tid + off]; shq[tid] += shq[tid + off]; }
        __syncthreads();
    }
    if (tid == 0) {
        const float mu  = shs[0] / (float)cnt;
        const float var = shq[0] / (float)cnt - mu * mu;
        meanOut[n] = mu;
        rstdOut[n] = rsqrtf(var + 1e-5f);
    }
}

__global__ __launch_bounds__(256)
void bn_apply_kernel(half_t* __restrict__ x, const float* __restrict__ mean,
                     const float* __restrict__ rstd, const float* __restrict__ g,
                     const float* __restrict__ bb, int TT, long total)
{
    const long i = (long)blockIdx.x * 256 + threadIdx.x;
    if (i >= total) return;
    const int n = (int)((i / (long)(TT * 64)) % 2000);
    const float v = (float)x[i];
    x[i] = (half_t)((v - mean[n]) * rstd[n] * g[n] + bb[n]);
}

// ---------------------------------------------------------------------------
// Final linear 896->1 + softplus; one wave32 per (b,n) row.
// ---------------------------------------------------------------------------
__global__ __launch_bounds__(256)
void linear_softplus_kernel(const half_t* __restrict__ h, const float* __restrict__ lw,
                            const float* __restrict__ lb, float* __restrict__ out)
{
    const int gid  = blockIdx.x * 256 + threadIdx.x;
    const int wave = gid >> 5;
    const int lane = threadIdx.x & 31;
    if (wave >= 16000) return;
    const half_t* row = h + (long)wave * 896;
    float s = 0.f;
    for (int i = lane; i < 896; i += 32)
        s += (float)row[i] * lw[i];
    #pragma unroll
    for (int off = 16; off > 0; off >>= 1)
        s += __shfl_xor(s, off, 32);
    if (lane == 0) {
        const float xx = s + lb[0];
        float sp;
        if (xx > 20.f)       sp = xx;
        else if (xx < -20.f) sp = expf(xx);
        else                 sp = log1pf(expf(xx));
        out[wave] = sp;
    }
}

// ---------------------------------------------------------------------------
// Launch: params arrive as pytree leaves in sorted-key order after
// x, edge_index, weight_index:
//  3:lin_b 4:lin_w 5:lt_b1 6:lt_b2 7:lt_b3 8:lt_w1 9:lt_w2 10:lt_w3
//  st1: 11:bn_b 12:bn_g 13:cheb_b 14:cheb_w 15..17:t1_b1..3 18..20:t1_w1..3
//       21..23:t2_b1..3 24..26:t2_w1..3
//  st2: 27:bn_b 28:bn_g 29:cheb_b 30:cheb_w 31..33:t1_b1..3 34..36:t1_w1..3
//       37..39:t2_b1..3 40..42:t2_w1..3
// ---------------------------------------------------------------------------
extern "C" void kernel_launch(void* const* d_in, const int* in_sizes, int n_in,
                              void* d_out, int out_size, void* d_ws, size_t ws_size,
                              hipStream_t stream)
{
    (void)in_sizes; (void)n_in; (void)out_size; (void)ws_size;
    auto F = [&](int i) { return (const float*)d_in[i]; };

    const float* x = F(0);               // (8,24,2000,2) f32
    // d_in[1] edge_index, d_in[2] weight_index: dead (ChebConv K=1).

    char*   ws    = (char*)d_ws;
    half_t* buf0  = (half_t*)ws;                               // <= 40.96 MB used
    half_t* buf1  = (half_t*)(ws + (size_t)41 * 1024 * 1024);  // <= 32.77 MB used
    float*  mean0 = (float*)(ws + (size_t)75 * 1024 * 1024);
    float*  rstd0 = mean0 + 2048;
    float*  mean1 = rstd0 + 2048;
    float*  rstd1 = mean1 + 2048;

    // ---- st1 ----
    // tconv1: (8,24,2000,2)f32 -> (8*2000, 22, 16)f16 ; rows=352000 -> 22000 tiles
    tconv_gate_kernel<2, 16, 16, 24, 22, true><<<22000 / 8, 256, 0, stream>>>(
        x, F(18), F(15), F(19), F(16), F(20), F(17), buf0, 22000, 0);
    // cheb 16->16 + relu
    cheb_relu_kernel<<<22000 / 8, 256, 0, stream>>>(buf0, F(14), F(13), buf1, 22000);
    // tconv2: 16->64, T 22->20 ; rows=320000 -> 20000 tiles
    tconv_gate_kernel<16, 64, 64, 22, 20, false><<<20000 / 8, 256, 0, stream>>>(
        buf1, F(24), F(21), F(25), F(22), F(26), F(23), buf0, 20000, 0);
    // BN over node channel
    bn_stats_kernel<<<2000, 256, 0, stream>>>(buf0, mean0, rstd0, 20);
    bn_apply_kernel<<<80000, 256, 0, stream>>>(buf0, mean0, rstd0, F(12), F(11),
                                               20, (long)8 * 2000 * 20 * 64);

    // ---- st2 ----
    // tconv1: 64->16, T 20->18 ; rows=288000 -> 18000 tiles (L=192, 6 K-chunks)
    tconv_gate_kernel<64, 16, 16, 20, 18, false><<<18000 / 8, 256, 0, stream>>>(
        buf0, F(34), F(31), F(35), F(32), F(36), F(33), buf1, 18000, 0);
    cheb_relu_kernel<<<18000 / 8, 256, 0, stream>>>(buf1, F(30), F(29), buf0, 18000);
    // tconv2: 16->64, T 18->16 ; rows=256000 -> 16000 tiles
    tconv_gate_kernel<16, 64, 64, 18, 16, false><<<16000 / 8, 256, 0, stream>>>(
        buf0, F(40), F(37), F(41), F(38), F(42), F(39), buf1, 16000, 0);
    bn_stats_kernel<<<2000, 256, 0, stream>>>(buf1, mean1, rstd1, 16);
    bn_apply_kernel<<<64000, 256, 0, stream>>>(buf1, mean1, rstd1, F(28), F(27),
                                               16, (long)8 * 2000 * 16 * 64);

    // ---- last tconv: 64->64, T 16->14 ; rows=224000 -> 14000 tiles ----
    // split into two 32-column halves to keep static LDS (3*32*192*2 B) < 64 KB
    tconv_gate_kernel<64, 64, 32, 16, 14, false><<<14000 / 8, 256, 0, stream>>>(
        buf1, F(8), F(5), F(9), F(6), F(10), F(7), buf0, 14000, 0);
    tconv_gate_kernel<64, 64, 32, 16, 14, false><<<14000 / 8, 256, 0, stream>>>(
        buf1, F(8), F(5), F(9), F(6), F(10), F(7), buf0, 14000, 32);

    // ---- linear 896->1 + softplus: 16000 wave-rows, 2000 blocks ----
    linear_softplus_kernel<<<2000, 256, 0, stream>>>(buf0, F(4), F(3), (float*)d_out);
}
